// LlamaAttention_11673721111171
// MI455X (gfx1250) — compile-verified, matrix operand order checked by reading, not run on hardware
//
#include <hip/hip_runtime.h>
#include <hip/hip_bf16.h>

// ---------------------------------------------------------------------------
// Problem constants (from reference): B=4, S=8192, HID=1024, H=D=32
// ---------------------------------------------------------------------------
#define MROWS  32768        // B*S tokens
#define HID    1024
#define HEADS  32
#define OUT_ELEMS (MROWS * HID)

typedef __attribute__((ext_vector_type(16))) __bf16 v16bf;
typedef __attribute__((ext_vector_type(8)))  float  v8f;
typedef __attribute__((ext_vector_type(4)))  __bf16 v4bf;

union Frag  { v16bf v; uint4 u[2]; };
union FragH { v16bf v; __bf16 h[16]; };
union Pk8   { __bf16 h[8]; uint4 u; };

// ---------------------------------------------------------------------------
// fp32 -> bf16 conversion (vectorized)
// ---------------------------------------------------------------------------
__global__ void cvt_f32_to_bf16(const float* __restrict__ src,
                                __bf16* __restrict__ dst, int n) {
    int i = (blockIdx.x * blockDim.x + threadIdx.x) * 4;
    if (i < n) {
        float4 v = *(const float4*)(src + i);
        v4bf o;
        o.x = (__bf16)v.x; o.y = (__bf16)v.y;
        o.z = (__bf16)v.z; o.w = (__bf16)v.w;
        *(v4bf*)(dst + i) = o;
    }
}

// ---------------------------------------------------------------------------
// VIB masks + KL scalar (single wave)
// ---------------------------------------------------------------------------
__global__ void mask_kl_kernel(const float* __restrict__ mu_q,
                               const float* __restrict__ sig_q,
                               const float* __restrict__ mu_kv,
                               const float* __restrict__ sig_kv,
                               const float* __restrict__ eps_q,
                               const float* __restrict__ eps_kv,
                               float* __restrict__ mask_q,
                               float* __restrict__ mask_kv,
                               float* __restrict__ kl_out) {
    int l = threadIdx.x;  // 32 lanes
    float mq = mu_q[l],  sq = sig_q[l];
    float mk = mu_kv[l], sk = sig_kv[l];
    mask_q[l]  = 1.0f / (1.0f + __expf(-(mq + eps_q[l]  * sq)));
    mask_kv[l] = 1.0f / (1.0f + __expf(-(mk + eps_kv[l] * sk)));
    float t = (1.0f + sq - mq * mq - __expf(sq))
            + (1.0f + sk - mk * mk - __expf(sk));
    #pragma unroll
    for (int off = 16; off > 0; off >>= 1) t += __shfl_xor(t, off, 32);
    if (l == 0) *kl_out = -0.5f * (t / (float)HEADS);
}

// ---------------------------------------------------------------------------
// Tiled bf16 WMMA GEMM, transposed accumulate:  D = W_tile x X_tile^T
//   => lane owns one output row m, VGPR index r runs over 8 consecutive cols.
// Staging: global_load_async_to_lds_b128 (ASYNCcnt), LDS double-buffered.
// Block: 256 threads (8 waves), tile 128x128, K-step 64.
// ---------------------------------------------------------------------------
#define BM 128
#define BN 128
#define BK 64
#define LDT 72   // LDS row stride in bf16 (144 B: 16B-aligned, bank-spread)

__global__ __launch_bounds__(256)
void gemm_bf16_wmma(const __bf16* __restrict__ A,   // [M,K]
                    const __bf16* __restrict__ W,   // [N,K]
                    void* __restrict__ Cout,
                    const float* __restrict__ colscale,  // nullptr -> 1.0
                    int M, int N, int K,
                    int out_f32, int permute_v) {
    __shared__ __bf16 sA[2][BM * LDT];
    __shared__ __bf16 sW[2][BN * LDT];

    const int nblk = N / BN;
    const int bm = blockIdx.x / nblk;
    const int bn = blockIdx.x % nblk;
    const int m0 = bm * BM, n0 = bn * BN;

    const int tid  = threadIdx.x;
    const int lane = tid & 31;
    const int wave = tid >> 5;
    const int wm   = wave & 3;   // 0..3 along M
    const int wn   = wave >> 2;  // 0..1 along N
    const int l15  = lane & 15;
    const int hi   = lane >> 4;

    // async stage of one 128x64 tile of A and one of W into LDS buffer `buf`
    auto stage = [&](int k0, int buf) {
        #pragma unroll
        for (int i = 0; i < 4; ++i) {
            int c   = tid + 256 * i;
            int row = c >> 3;
            int col = (c & 7) * 8;
            unsigned la = (unsigned)(size_t)&sA[buf][row * LDT + col];
            unsigned long long ga =
                (unsigned long long)(size_t)&A[(size_t)(m0 + row) * K + k0 + col];
            asm volatile("global_load_async_to_lds_b128 %0, %1, off"
                         :: "v"(la), "v"(ga) : "memory");
            unsigned lw = (unsigned)(size_t)&sW[buf][row * LDT + col];
            unsigned long long gw =
                (unsigned long long)(size_t)&W[(size_t)(n0 + row) * K + k0 + col];
            asm volatile("global_load_async_to_lds_b128 %0, %1, off"
                         :: "v"(lw), "v"(gw) : "memory");
        }
    };

    v8f acc[2][4];
    const v8f vz = {0.f, 0.f, 0.f, 0.f, 0.f, 0.f, 0.f, 0.f};
    #pragma unroll
    for (int f = 0; f < 2; ++f)
        #pragma unroll
        for (int g = 0; g < 4; ++g) acc[f][g] = vz;

    const int nsteps = K / BK;
    stage(0, 0);
    for (int t = 0; t < nsteps; ++t) {
        asm volatile("s_wait_asynccnt 0x0" ::: "memory");
        __syncthreads();
        if (t + 1 < nsteps) stage((t + 1) * BK, (t + 1) & 1);

        const __bf16* cA = sA[t & 1];
        const __bf16* cW = sW[t & 1];
        #pragma unroll
        for (int kk = 0; kk < BK; kk += 32) {
            Frag fx[2], fw[4];
            #pragma unroll
            for (int f = 0; f < 2; ++f) {
                int row = wm * 32 + f * 16 + l15;
                fx[f].u[0] = *(const uint4*)&cA[row * LDT + kk + hi * 8];
                fx[f].u[1] = *(const uint4*)&cA[row * LDT + kk + 16 + hi * 8];
            }
            #pragma unroll
            for (int g = 0; g < 4; ++g) {
                int row = wn * 64 + g * 16 + l15;
                fw[g].u[0] = *(const uint4*)&cW[row * LDT + kk + hi * 8];
                fw[g].u[1] = *(const uint4*)&cW[row * LDT + kk + 16 + hi * 8];
            }
            // transposed accumulate: D[n,m] — A-op = W rows, B-op = X rows
            #pragma unroll
            for (int f = 0; f < 2; ++f)
                #pragma unroll
                for (int g = 0; g < 4; ++g)
                    acc[f][g] = __builtin_amdgcn_wmma_f32_16x16x32_bf16(
                        false, fw[g].v, false, fx[f].v,
                        (short)0, acc[f][g], false, false);
        }
        __syncthreads();
    }

    // epilogue: lane holds row m = ...+l15; r runs over 8 consecutive columns
    #pragma unroll
    for (int f = 0; f < 2; ++f) {
        const int m = m0 + wm * 32 + f * 16 + l15;
        #pragma unroll
        for (int g = 0; g < 4; ++g) {
            const int nb = n0 + wn * 64 + g * 16 + 8 * hi;  // 8 consecutive cols
            float s[8];
            if (colscale) {
                float4 s0 = *(const float4*)&colscale[nb & 31];
                float4 s1 = *(const float4*)&colscale[(nb & 31) + 4];
                s[0] = s0.x; s[1] = s0.y; s[2] = s0.z; s[3] = s0.w;
                s[4] = s1.x; s[5] = s1.y; s[6] = s1.z; s[7] = s1.w;
            } else {
                #pragma unroll
                for (int r = 0; r < 8; ++r) s[r] = 1.0f;
            }
            if (permute_v) {  // V: store V^T within each token (scatter)
                #pragma unroll
                for (int r = 0; r < 8; ++r) {
                    int n = nb + r;
                    size_t nn = (size_t)((n & 31) * 32 + (n >> 5));
                    ((__bf16*)Cout)[(size_t)m * N + nn] =
                        (__bf16)(acc[f][g][r] * s[r]);
                }
            } else if (out_f32) {
                float4 lo, hi4;
                lo.x  = acc[f][g][0] * s[0]; lo.y  = acc[f][g][1] * s[1];
                lo.z  = acc[f][g][2] * s[2]; lo.w  = acc[f][g][3] * s[3];
                hi4.x = acc[f][g][4] * s[4]; hi4.y = acc[f][g][5] * s[5];
                hi4.z = acc[f][g][6] * s[6]; hi4.w = acc[f][g][7] * s[7];
                float* p = (float*)Cout + (size_t)m * N + nb;
                *(float4*)p       = lo;
                *(float4*)(p + 4) = hi4;
            } else {
                Pk8 pk;
                #pragma unroll
                for (int r = 0; r < 8; ++r)
                    pk.h[r] = (__bf16)(acc[f][g][r] * s[r]);
                *(uint4*)((__bf16*)Cout + (size_t)m * N + nb) = pk.u;
            }
        }
    }
}

// ---------------------------------------------------------------------------
// Per-token 32x32 head-vs-head attention. One wave per token, 8 tokens/block.
// scores^T = K x Q^T  (4 WMMA) -> softmax over t via shfl_xor(16) -> the
// normalized weights land exactly in the B-operand fragment layout for
// o^T = V^T x w^T (4 WMMA). No LDS, contiguous 16B output stores.
// ---------------------------------------------------------------------------
__global__ __launch_bounds__(256)
void attn_token_kernel(const __bf16* __restrict__ Q,
                       const __bf16* __restrict__ Kp,
                       const __bf16* __restrict__ VT,   // [d, t] per token
                       __bf16* __restrict__ Attn) {
    const int lane = threadIdx.x & 31;
    const int wave = threadIdx.x >> 5;
    const int tok  = blockIdx.x * 8 + wave;
    const int l15  = lane & 15;
    const int hi   = lane >> 4;

    const __bf16* q  = Q  + (size_t)tok * HID;
    const __bf16* k  = Kp + (size_t)tok * HID;
    const __bf16* vt = VT + (size_t)tok * HID;

    Frag ak[2], bq[2], av[2];
    #pragma unroll
    for (int f = 0; f < 2; ++f) {
        int row = f * 16 + l15;
        ak[f].u[0] = *(const uint4*)&k[row * 32 + hi * 8];
        ak[f].u[1] = *(const uint4*)&k[row * 32 + 16 + hi * 8];
        bq[f].u[0] = *(const uint4*)&q[row * 32 + hi * 8];
        bq[f].u[1] = *(const uint4*)&q[row * 32 + 16 + hi * 8];
        av[f].u[0] = *(const uint4*)&vt[row * 32 + hi * 8];
        av[f].u[1] = *(const uint4*)&vt[row * 32 + 16 + hi * 8];
    }

    const v8f vz = {0.f, 0.f, 0.f, 0.f, 0.f, 0.f, 0.f, 0.f};
    // scores^T[t,h]: frag (u,g): lane -> h = 16g+l15, vgpr r -> t = 16u+8hi+r
    v8f sacc[2][2];
    #pragma unroll
    for (int u = 0; u < 2; ++u)
        #pragma unroll
        for (int g = 0; g < 2; ++g)
            sacc[u][g] = __builtin_amdgcn_wmma_f32_16x16x32_bf16(
                false, ak[u].v, false, bq[g].v, (short)0, vz, false, false);

    // softmax over t per column h; lanes l and l^16 split the 32 t-values
    const float inv_sqrt_d = 0.17677669529663687f;  // 1/sqrt(32)
    FragH vb[2];
    #pragma unroll
    for (int g = 0; g < 2; ++g) {
        float e[2][8];
        float mx = -3.0e38f;
        #pragma unroll
        for (int u = 0; u < 2; ++u)
            #pragma unroll
            for (int r = 0; r < 8; ++r) {
                float v = sacc[u][g][r] * inv_sqrt_d;
                e[u][r] = v;
                mx = fmaxf(mx, v);
            }
        mx = fmaxf(mx, __shfl_xor(mx, 16, 32));
        float sum = 0.f;
        #pragma unroll
        for (int u = 0; u < 2; ++u)
            #pragma unroll
            for (int r = 0; r < 8; ++r) {
                e[u][r] = __expf(e[u][r] - mx);
                sum += e[u][r];
            }
        sum += __shfl_xor(sum, 16, 32);
        float rs = 1.0f / sum;
        // repack: fragment element j -> t = (j<8 ? 8hi+j : 16+8hi+(j-8))
        #pragma unroll
        for (int u = 0; u < 2; ++u)
            #pragma unroll
            for (int r = 0; r < 8; ++r)
                vb[g].h[u * 8 + r] = (__bf16)(e[u][r] * rs);
    }

    // o^T[d,h] = sum_t VT[d,t] * w[h,t]; frag (f,g): lane -> h = 16g+l15,
    // vgpr r -> d = 16f+8hi+r  => contiguous 8 bf16 per lane in out[h*32+d]
    __bf16* out = Attn + (size_t)tok * HID;
    #pragma unroll
    for (int f = 0; f < 2; ++f)
        #pragma unroll
        for (int g = 0; g < 2; ++g) {
            v8f o = __builtin_amdgcn_wmma_f32_16x16x32_bf16(
                false, av[f].v, false, vb[g].v, (short)0, vz, false, false);
            Pk8 pk;
            #pragma unroll
            for (int r = 0; r < 8; ++r) pk.h[r] = (__bf16)o[r];
            *(uint4*)&out[(g * 16 + l15) * 32 + f * 16 + 8 * hi] = pk.u;
        }
}

// ---------------------------------------------------------------------------
// Host launcher
// ---------------------------------------------------------------------------
extern "C" void kernel_launch(void* const* d_in, const int* in_sizes, int n_in,
                              void* d_out, int out_size, void* d_ws, size_t ws_size,
                              hipStream_t stream) {
    const float* X      = (const float*)d_in[0];
    const float* wq     = (const float*)d_in[1];
    const float* wk     = (const float*)d_in[2];
    const float* wv     = (const float*)d_in[3];
    const float* wo     = (const float*)d_in[4];
    const float* mu_q   = (const float*)d_in[5];
    const float* sig_q  = (const float*)d_in[6];
    const float* mu_kv  = (const float*)d_in[7];
    const float* sig_kv = (const float*)d_in[8];
    const float* eps_q  = (const float*)d_in[9];
    const float* eps_kv = (const float*)d_in[10];

    char* ws = (char*)d_ws;
    size_t off = 0;
    auto take = [&](size_t bytes) {
        void* p = ws + off;
        off += (bytes + 255) & ~(size_t)255;
        return p;
    };
    __bf16* Xbf   = (__bf16*)take((size_t)OUT_ELEMS * 2);
    __bf16* Wqbf  = (__bf16*)take((size_t)HID * HID * 2);
    __bf16* Wkbf  = (__bf16*)take((size_t)HID * HID * 2);
    __bf16* Wvbf  = (__bf16*)take((size_t)HID * HID * 2);
    __bf16* Wobf  = (__bf16*)take((size_t)HID * HID * 2);
    __bf16* Qbf   = (__bf16*)take((size_t)OUT_ELEMS * 2);
    __bf16* Kbf   = (__bf16*)take((size_t)OUT_ELEMS * 2);
    __bf16* VTbf  = (__bf16*)take((size_t)OUT_ELEMS * 2);
    __bf16* Attbf = (__bf16*)take((size_t)OUT_ELEMS * 2);
    float*  maskq = (float*)take(32 * sizeof(float));
    float*  maskk = (float*)take(32 * sizeof(float));
    (void)ws_size; (void)in_sizes; (void)n_in; (void)out_size;

    // 1) bf16 conversions
    {
        int n = OUT_ELEMS;
        cvt_f32_to_bf16<<<(n / 4 + 255) / 256, 256, 0, stream>>>(X, Xbf, n);
        int w = HID * HID;
        int wb = (w / 4 + 255) / 256;
        cvt_f32_to_bf16<<<wb, 256, 0, stream>>>(wq, Wqbf, w);
        cvt_f32_to_bf16<<<wb, 256, 0, stream>>>(wk, Wkbf, w);
        cvt_f32_to_bf16<<<wb, 256, 0, stream>>>(wv, Wvbf, w);
        cvt_f32_to_bf16<<<wb, 256, 0, stream>>>(wo, Wobf, w);
    }

    // 2) masks + KL (kl written at d_out[OUT_ELEMS])
    mask_kl_kernel<<<1, 32, 0, stream>>>(mu_q, sig_q, mu_kv, sig_kv,
                                         eps_q, eps_kv, maskq, maskk,
                                         (float*)d_out + OUT_ELEMS);

    // 3) projections
    const int gemm_blocks = (MROWS / BM) * (HID / BN);   // 256 * 8 = 2048
    gemm_bf16_wmma<<<gemm_blocks, 256, 0, stream>>>(
        Xbf, Wqbf, Qbf, maskq, MROWS, HID, HID, /*f32*/0, /*perm*/0);
    gemm_bf16_wmma<<<gemm_blocks, 256, 0, stream>>>(
        Xbf, Wkbf, Kbf, maskk, MROWS, HID, HID, 0, 0);
    gemm_bf16_wmma<<<gemm_blocks, 256, 0, stream>>>(
        Xbf, Wvbf, VTbf, maskk, MROWS, HID, HID, 0, /*perm V^T*/1);

    // 4) per-token attention (8 tokens per 256-thread block)
    attn_token_kernel<<<MROWS / 8, 256, 0, stream>>>(Qbf, Kbf, VTbf, Attbf);

    // 5) output projection (fp32 straight to d_out)
    gemm_bf16_wmma<<<gemm_blocks, 256, 0, stream>>>(
        Attbf, Wobf, d_out, nullptr, MROWS, HID, HID, /*f32*/1, 0);
}
